// MaskMedNet_64201171141229
// MI455X (gfx1250) — matrix operand order repaired and checked by reading, not compiled, not run on hardware
//
#include <hip/hip_runtime.h>

// ---------------------------------------------------------------------------
// FSAD (3D deformable attention) for MI455X / gfx1250, wave32 + WMMA bf16.
//
// Memory-bound problem (~350 MB of traffic, ~5.4 GFLOP) -> bf16 everywhere for
// matrix operands (val0 intermediate = 100 MB bf16, L2-resident), fp32 accum.
//
// Pipeline (all on `stream`, intermediates in d_ws, ~152 MB needed):
//   0. convert weights -> bf16 [M,K]; transpose inputs -> bf16 [N,K]
//   1. q   = Wq  @ Qf                  -> bf16 [4096,192]   (WMMA)
//   2. off = tanh(Woff @ q + boff)*.35 -> f32  [4096,384]   (WMMA + epilogue)
//   3. aw  = softmax16(Wwt @ q + bwt)  -> f32  [4096,128]   (WMMA + shfl_xor)
//   4. val_s = Wv_s @ v_s              -> bf16 [N_s,192]    (WMMA, s=0..3)
//   5. deformable trilinear gather     -> acc bf16 [4096,192]
//   6. pre = Wout @ acc                -> f32  [128,4096]   (WMMA)
//   7. InstanceNorm + affine           -> d_out f32 [128,4096]
// ---------------------------------------------------------------------------

typedef __bf16 bf16_t;
typedef bf16_t v16bf __attribute__((ext_vector_type(16)));
typedef float  v8f   __attribute__((ext_vector_type(8)));

__device__ __forceinline__ unsigned short bfbits(float f) {
  unsigned u = __builtin_bit_cast(unsigned, f);
  u += 0x7fffu + ((u >> 16) & 1u);            // round-to-nearest-even
  return (unsigned short)(u >> 16);
}
__device__ __forceinline__ bf16_t f2bf(float f) {
  unsigned short h = bfbits(f);
  return __builtin_bit_cast(bf16_t, h);
}
__device__ __forceinline__ float bflo(unsigned u) {
  return __builtin_bit_cast(float, u << 16);
}
__device__ __forceinline__ float bfhi(unsigned u) {
  return __builtin_bit_cast(float, u & 0xffff0000u);
}

// ---------------------------------------------------------------------------
// Elementwise fp32 -> bf16 (weights, layout preserved).
// ---------------------------------------------------------------------------
__global__ void cvt_k(const float* __restrict__ in, bf16_t* __restrict__ out, int n) {
  int i = blockIdx.x * 256 + threadIdx.x;
  if (i < n) out[i] = f2bf(in[i]);
}

// ---------------------------------------------------------------------------
// Tiled transpose + convert: fp32 [K,N] row-major -> bf16 [N,K].
// 32x32 tile via LDS; coalesced b32 loads, packed b64 bf16 stores.
// Requires K%32==0, N%32==0 (all shapes satisfy).
// ---------------------------------------------------------------------------
__global__ void tconv_k(const float* __restrict__ In, bf16_t* __restrict__ Out,
                        int K, int N) {
  __shared__ float T[32][33];
  const unsigned n0 = blockIdx.x * 32, k0 = blockIdx.y * 32;
  const int t = threadIdx.x;                       // 256 threads
#pragma unroll
  for (int r = 0; r < 4; ++r) {
    int id = t + r * 256;
    int kk = id >> 5, n = id & 31;
    T[kk][n] = In[(size_t)(k0 + kk) * N + n0 + n];
  }
  __syncthreads();
  {
    int n = t >> 3, kc = (t & 7) * 4;              // 4 consecutive k per thread
    uint2 st;
    st.x = (unsigned)bfbits(T[kc + 0][n]) | ((unsigned)bfbits(T[kc + 1][n]) << 16);
    st.y = (unsigned)bfbits(T[kc + 2][n]) | ((unsigned)bfbits(T[kc + 3][n]) << 16);
    *(uint2*)(Out + (size_t)(n0 + n) * K + k0 + kc) = st;
  }
}

// ---------------------------------------------------------------------------
// WMMA bf16 GEMM:  C[M,N] = A[M,K] * B[K,N]
//   A: bf16 row-major [M,K];  B: bf16 [N,K] (ld=K, i.e. B^T contiguous in K)
//   EPI 0: store bf16 column-major [N, ld=M]                    (q, val_s)
//   EPI 1: bias + tanh*0.35, fp32 column-major [N, ld=M]        (off)
//   EPI 2: bias + softmax over the 16-row M tile, fp32 [N,M]    (aw)
//   EPI 3: fp32 row-major [M, N]                                (pre-norm out)
// Block 128 = 4 waves; WG tile 16(M) x 128(N); each wave: 2 16x16 tiles,
// sharing one A fragment -> 2 v_wmma per K-step between barriers.
// Requires M%16==0, K%32==0, N%128==0. All indices fit in 32 bits.
// ---------------------------------------------------------------------------
template <int EPI>
__global__ void gemm_wmma_k(const bf16_t* __restrict__ A, const bf16_t* __restrict__ B,
                            void* __restrict__ Out, int M, int K, int N,
                            const float* __restrict__ bias) {
  __shared__ __align__(16) bf16_t As[16 * 40];    // 16 x 32 (+8 pad) bf16
  __shared__ __align__(16) bf16_t Bs[128 * 40];   // 128(n) x 32(k) (+8 pad)

  const int t    = threadIdx.x;
  const int wv   = t >> 5;
  const int lane = t & 31;
  const int m0   = blockIdx.y * 16;
  const unsigned n0 = blockIdx.x * 128u;

  v8f c0 = {}, c1 = {};

  // Staging coordinates (hoisted out of the K loop).
  const int an = t >> 2, ach = t & 3;             // B: 128 cols x 4 b128 chunks
  const int bm = t >> 2;                          // A: 16 rows x 4 chunks (t<64)

  for (int k0 = 0; k0 < K; k0 += 32) {
    if (t < 64)                                   // A tile: 64 b128 copies
      *(uint4*)&As[bm * 40 + ach * 8] =
          *(const uint4*)(A + (unsigned)(m0 + bm) * K + k0 + ach * 8);
#pragma unroll
    for (int r = 0; r < 4; ++r) {                 // B tile: 512 b128 copies
      int n = an + r * 32;
      *(uint4*)&Bs[n * 40 + ach * 8] =
          *(const uint4*)(B + (unsigned)(n0 + n) * K + k0 + ach * 8);
    }
    __syncthreads();

    // Fragments: m/n = lane&15, k = 16*(lane>>4) + element, same k order in A and B.
    v16bf a, b0, b1;
    {
      const int am = lane & 15, akb = (lane >> 4) * 16;
#pragma unroll
      for (int j = 0; j < 8; ++j) {
        a[j]     = As[am * 40 + akb + j];
        a[j + 8] = As[am * 40 + akb + 8 + j];
      }
      const int bn  = wv * 32 + (lane & 15);
      const int bkb = (lane >> 4) * 16;
#pragma unroll
      for (int j = 0; j < 16; ++j) {
        b0[j] = Bs[bn * 40 + bkb + j];
        b1[j] = Bs[(bn + 16) * 40 + bkb + j];
      }
    }
    c0 = __builtin_amdgcn_wmma_f32_16x16x32_bf16(false, a, false, b0, (short)0, c0,
                                                 false, false);
    c1 = __builtin_amdgcn_wmma_f32_16x16x32_bf16(false, a, false, b1, (short)0, c1,
                                                 false, false);
    __syncthreads();
  }

  // Epilogue. D layout: lane -> col = lane&15, rows (lane>>4)*8 .. +7.
  const int col = lane & 15;
  const int mb  = (lane >> 4) << 3;
#pragma unroll
  for (int u = 0; u < 2; ++u) {
    const unsigned n = n0 + wv * 32 + u * 16 + col;
    float tv[8];
#pragma unroll
    for (int v = 0; v < 8; ++v) tv[v] = (u == 0) ? c0[v] : c1[v];

    if (EPI == 0) {                               // bf16 [N, M], one b128 store
      uint4 st;
      st.x = (unsigned)bfbits(tv[0]) | ((unsigned)bfbits(tv[1]) << 16);
      st.y = (unsigned)bfbits(tv[2]) | ((unsigned)bfbits(tv[3]) << 16);
      st.z = (unsigned)bfbits(tv[4]) | ((unsigned)bfbits(tv[5]) << 16);
      st.w = (unsigned)bfbits(tv[6]) | ((unsigned)bfbits(tv[7]) << 16);
      *(uint4*)((bf16_t*)Out + (unsigned)(n * M) + m0 + mb) = st;
    } else if (EPI == 1) {                        // bias + tanh*0.35, f32 [N,M]
      float* o = (float*)Out + (unsigned)(n * M) + m0 + mb;
#pragma unroll
      for (int v = 0; v < 8; ++v)
        o[v] = tanhf(tv[v] + bias[m0 + mb + v]) * 0.35f;
    } else if (EPI == 2) {                        // bias + softmax over 16 rows
      float mx = -3.0e38f;
#pragma unroll
      for (int v = 0; v < 8; ++v) { tv[v] += bias[m0 + mb + v]; mx = fmaxf(mx, tv[v]); }
      mx = fmaxf(mx, __shfl_xor(mx, 16));         // rows 8..15 live in lane^16
      float e[8], s = 0.f;
#pragma unroll
      for (int v = 0; v < 8; ++v) { e[v] = __expf(tv[v] - mx); s += e[v]; }
      s += __shfl_xor(s, 16);
      const float inv = 1.0f / s;
      float* o = (float*)Out + (unsigned)(n * M) + m0 + mb;
#pragma unroll
      for (int v = 0; v < 8; ++v) o[v] = e[v] * inv;
    } else {                                      // EPI 3: f32 row-major [M, N]
      float* o = (float*)Out;
#pragma unroll
      for (int v = 0; v < 8; ++v) o[(unsigned)((m0 + mb + v) * N) + n] = tv[v];
    }
  }
}

// ---------------------------------------------------------------------------
// Deformable trilinear gather. 1 thread = (voxel i, head h).
// val_s layout [N_s, 192] bf16 -> 24 channels of a head are 48 contiguous
// bytes = 3 x b128 per corner. acc stored bf16 [4096, 192].
// ---------------------------------------------------------------------------
__global__ void sampler_k(const float* __restrict__ off,  // [4096,384]
                          const float* __restrict__ aw,   // [4096,128]
                          const bf16_t* __restrict__ va0, const bf16_t* __restrict__ va1,
                          const bf16_t* __restrict__ va2, const bf16_t* __restrict__ va3,
                          bf16_t* __restrict__ acc) {
  const int t = blockIdx.x * blockDim.x + threadIdx.x;   // 0 .. 32767
  const int h = t & 7;
  const int i = t >> 3;
  const int xi = i & 15, yi = (i >> 4) & 15, zi = i >> 8;
  const float bgx = -1.f + (2.f / 15.f) * xi;
  const float bgy = -1.f + (2.f / 15.f) * yi;
  const float bgz = -1.f + (2.f / 15.f) * zi;

  const float* offp = off + (unsigned)(i * 384) + h * 48;
  const float* awp  = aw  + (unsigned)(i * 128) + h * 16;

  float av[24];
#pragma unroll
  for (int c = 0; c < 24; ++c) av[c] = 0.f;

  const bf16_t* vals[4] = {va0, va1, va2, va3};
  const int dims[4] = {64, 32, 16, 8};

  for (int s = 0; s < 4; ++s) {
    const bf16_t* val = vals[s];
    const int Ds = dims[s];
    const float fs = (float)Ds;
    auto corner = [&](int zz, int yy, int xx, float w) {
      const uint4* p =
          (const uint4*)(val + (unsigned)(((zz * Ds + yy) * Ds + xx) * 192) + h * 24);
      uint4 q0 = p[0], q1 = p[1], q2 = p[2];
      unsigned uu[12] = {q0.x, q0.y, q0.z, q0.w, q1.x, q1.y,
                         q1.z, q1.w, q2.x, q2.y, q2.z, q2.w};
#pragma unroll
      for (int j = 0; j < 12; ++j) {
        av[2 * j]     += w * bflo(uu[j]);
        av[2 * j + 1] += w * bfhi(uu[j]);
      }
    };
    for (int p = 0; p < 4; ++p) {
      const int o = (s * 4 + p) * 3;
      const float gx = bgx + offp[o], gy = bgy + offp[o + 1], gz = bgz + offp[o + 2];
      const float a_w = awp[s * 4 + p];
      // align_corners=False unnormalize + border clamp
      float ix = fminf(fmaxf(((gx + 1.f) * fs - 1.f) * 0.5f, 0.f), fs - 1.f);
      float iy = fminf(fmaxf(((gy + 1.f) * fs - 1.f) * 0.5f, 0.f), fs - 1.f);
      float iz = fminf(fmaxf(((gz + 1.f) * fs - 1.f) * 0.5f, 0.f), fs - 1.f);
      float x0f = floorf(ix), y0f = floorf(iy), z0f = floorf(iz);
      float wx = ix - x0f, wy = iy - y0f, wz = iz - z0f;
      int x0 = (int)x0f, y0 = (int)y0f, z0 = (int)z0f;
      int x1 = min(x0 + 1, Ds - 1), y1 = min(y0 + 1, Ds - 1), z1 = min(z0 + 1, Ds - 1);
      corner(z0, y0, x0, a_w * (1 - wz) * (1 - wy) * (1 - wx));
      corner(z0, y0, x1, a_w * (1 - wz) * (1 - wy) * wx);
      corner(z0, y1, x0, a_w * (1 - wz) * wy * (1 - wx));
      corner(z0, y1, x1, a_w * (1 - wz) * wy * wx);
      corner(z1, y0, x0, a_w * wz * (1 - wy) * (1 - wx));
      corner(z1, y0, x1, a_w * wz * (1 - wy) * wx);
      corner(z1, y1, x0, a_w * wz * wy * (1 - wx));
      corner(z1, y1, x1, a_w * wz * wy * wx);
    }
  }

  bf16_t* ap = acc + (unsigned)(i * 192) + h * 24;
#pragma unroll
  for (int q = 0; q < 3; ++q) {
    uint4 st;
    st.x = (unsigned)bfbits(av[8 * q + 0]) | ((unsigned)bfbits(av[8 * q + 1]) << 16);
    st.y = (unsigned)bfbits(av[8 * q + 2]) | ((unsigned)bfbits(av[8 * q + 3]) << 16);
    st.z = (unsigned)bfbits(av[8 * q + 4]) | ((unsigned)bfbits(av[8 * q + 5]) << 16);
    st.w = (unsigned)bfbits(av[8 * q + 6]) | ((unsigned)bfbits(av[8 * q + 7]) << 16);
    ((uint4*)ap)[q] = st;
  }
}

// ---------------------------------------------------------------------------
// InstanceNorm3d (biased variance) + affine. One WG per channel (row of 4096).
// ---------------------------------------------------------------------------
__global__ void inorm_k(const float* __restrict__ x, const float* __restrict__ gamma,
                        const float* __restrict__ beta, float* __restrict__ out) {
  const int c = blockIdx.x;
  const int t = threadIdx.x;
  const float* row = x + (unsigned)(c * 4096);
  float s = 0.f, s2 = 0.f;
  for (int j = t; j < 4096; j += 256) {
    float v = row[j];
    s += v; s2 += v * v;
  }
  __shared__ float sh1[256], sh2[256];
  sh1[t] = s; sh2[t] = s2;
  __syncthreads();
  for (int st = 128; st > 0; st >>= 1) {
    if (t < st) { sh1[t] += sh1[t + st]; sh2[t] += sh2[t + st]; }
    __syncthreads();
  }
  const float mu  = sh1[0] * (1.f / 4096.f);
  const float var = sh2[0] * (1.f / 4096.f) - mu * mu;
  const float rs  = rsqrtf(var + 1e-5f);
  const float g = gamma[c] * rs;
  const float b = beta[c] - mu * g;
  for (int j = t; j < 4096; j += 256)
    out[(unsigned)(c * 4096) + j] = row[j] * g + b;
}

// ---------------------------------------------------------------------------
extern "C" void kernel_launch(void* const* d_in, const int* in_sizes, int n_in,
                              void* d_out, int out_size, void* d_ws, size_t ws_size,
                              hipStream_t stream) {
  const float* qf   = (const float*)d_in[0];    // [128,4096]
  const float* v0f  = (const float*)d_in[1];    // [32, 262144]
  const float* v1f  = (const float*)d_in[2];    // [64, 32768]
  const float* v2f  = (const float*)d_in[3];    // [128, 4096]
  const float* v3f  = (const float*)d_in[4];    // [256, 512]
  const float* Wq   = (const float*)d_in[5];
  const float* Wv0  = (const float*)d_in[6];
  const float* Wv1  = (const float*)d_in[7];
  const float* Wv2  = (const float*)d_in[8];
  const float* Wv3  = (const float*)d_in[9];
  const float* Woff = (const float*)d_in[10];
  const float* boff = (const float*)d_in[11];
  const float* Wwt  = (const float*)d_in[12];
  const float* bwt  = (const float*)d_in[13];
  const float* Wout = (const float*)d_in[14];
  const float* gam  = (const float*)d_in[15];
  const float* bet  = (const float*)d_in[16];
  (void)in_sizes; (void)n_in; (void)out_size; (void)ws_size;

  char* ws = (char*)d_ws;
  size_t o = 0;
  auto carve = [&](size_t bytes) { char* p = ws + o; o = (o + bytes + 255) & ~(size_t)255; return p; };
  // bf16 weights (layout preserved [M,K])
  bf16_t* WqB   = (bf16_t*)carve(192ull * 128 * 2);
  bf16_t* WoffB = (bf16_t*)carve(384ull * 192 * 2);
  bf16_t* WwtB  = (bf16_t*)carve(128ull * 192 * 2);
  bf16_t* Wv0B  = (bf16_t*)carve(192ull * 32 * 2);
  bf16_t* Wv1B  = (bf16_t*)carve(192ull * 64 * 2);
  bf16_t* Wv2B  = (bf16_t*)carve(192ull * 128 * 2);
  bf16_t* Wv3B  = (bf16_t*)carve(192ull * 256 * 2);
  bf16_t* WoutB = (bf16_t*)carve(128ull * 192 * 2);
  // bf16 transposed activations [N,K]
  bf16_t* qfT   = (bf16_t*)carve(4096ull * 128 * 2);
  bf16_t* v0T   = (bf16_t*)carve(262144ull * 32 * 2);
  bf16_t* v1T   = (bf16_t*)carve(32768ull * 64 * 2);
  bf16_t* v2T   = (bf16_t*)carve(4096ull * 128 * 2);
  bf16_t* v3T   = (bf16_t*)carve(512ull * 256 * 2);
  // intermediates
  bf16_t* q_bf  = (bf16_t*)carve(4096ull * 192 * 2);      // [4096,192]
  float*  offf  = (float*)carve(4096ull * 384 * 4);       // [4096,384]
  float*  awf   = (float*)carve(4096ull * 128 * 4);       // [4096,128]
  bf16_t* val0  = (bf16_t*)carve(262144ull * 192 * 2);    // [262144,192]
  bf16_t* val1  = (bf16_t*)carve(32768ull * 192 * 2);
  bf16_t* val2  = (bf16_t*)carve(4096ull * 192 * 2);
  bf16_t* val3  = (bf16_t*)carve(512ull * 192 * 2);
  bf16_t* accb  = (bf16_t*)carve(4096ull * 192 * 2);      // [4096,192]
  float*  pre   = (float*)carve(128ull * 4096 * 4);       // [128,4096]

  // 0a. weight conversions (tiny)
  cvt_k<<<dim3((192 * 128 + 255) / 256), 256, 0, stream>>>(Wq, WqB, 192 * 128);
  cvt_k<<<dim3((384 * 192 + 255) / 256), 256, 0, stream>>>(Woff, WoffB, 384 * 192);
  cvt_k<<<dim3((128 * 192 + 255) / 256), 256, 0, stream>>>(Wwt, WwtB, 128 * 192);
  cvt_k<<<dim3((192 * 32 + 255) / 256), 256, 0, stream>>>(Wv0, Wv0B, 192 * 32);
  cvt_k<<<dim3((192 * 64 + 255) / 256), 256, 0, stream>>>(Wv1, Wv1B, 192 * 64);
  cvt_k<<<dim3((192 * 128 + 255) / 256), 256, 0, stream>>>(Wv2, Wv2B, 192 * 128);
  cvt_k<<<dim3((192 * 256 + 255) / 256), 256, 0, stream>>>(Wv3, Wv3B, 192 * 256);
  cvt_k<<<dim3((128 * 192 + 255) / 256), 256, 0, stream>>>(Wout, WoutB, 128 * 192);
  // 0b. activation transposes fp32 [K,N] -> bf16 [N,K]
  tconv_k<<<dim3(128, 4), 256, 0, stream>>>(qf, qfT, 128, 4096);
  tconv_k<<<dim3(8192, 1), 256, 0, stream>>>(v0f, v0T, 32, 262144);
  tconv_k<<<dim3(1024, 2), 256, 0, stream>>>(v1f, v1T, 64, 32768);
  tconv_k<<<dim3(128, 4), 256, 0, stream>>>(v2f, v2T, 128, 4096);
  tconv_k<<<dim3(16, 8), 256, 0, stream>>>(v3f, v3T, 256, 512);

  // 1. q = Wq @ Qf                     M=192 K=128 N=4096
  gemm_wmma_k<0><<<dim3(32, 12), 128, 0, stream>>>(WqB, qfT, q_bf, 192, 128, 4096, nullptr);
  // 2. off = tanh(Woff @ q + boff)*.35 M=384 K=192 N=4096
  gemm_wmma_k<1><<<dim3(32, 24), 128, 0, stream>>>(WoffB, q_bf, offf, 384, 192, 4096, boff);
  // 3. aw = softmax16(Wwt @ q + bwt)   M=128 K=192 N=4096
  gemm_wmma_k<2><<<dim3(32, 8), 128, 0, stream>>>(WwtB, q_bf, awf, 128, 192, 4096, bwt);
  // 4. val_s = Wv_s @ v_s              M=192, K=C_s, N=N_s
  gemm_wmma_k<0><<<dim3(2048, 12), 128, 0, stream>>>(Wv0B, v0T, val0, 192, 32, 262144, nullptr);
  gemm_wmma_k<0><<<dim3(256, 12), 128, 0, stream>>>(Wv1B, v1T, val1, 192, 64, 32768, nullptr);
  gemm_wmma_k<0><<<dim3(32, 12), 128, 0, stream>>>(Wv2B, v2T, val2, 192, 128, 4096, nullptr);
  gemm_wmma_k<0><<<dim3(4, 12), 128, 0, stream>>>(Wv3B, v3T, val3, 192, 256, 512, nullptr);
  // 5. deformable gather -> acc
  sampler_k<<<128, 256, 0, stream>>>(offf, awf, val0, val1, val2, val3, accb);
  // 6. pre = Wout @ acc                M=128 K=192 N=4096, row-major f32
  gemm_wmma_k<3><<<dim3(32, 8), 128, 0, stream>>>(WoutB, accb, pre, 128, 192, 4096, nullptr);
  // 7. InstanceNorm + affine -> d_out
  inorm_k<<<128, 256, 0, stream>>>(pre, gam, bet, (float*)d_out);
}
